// Graph2Vec_40948218200427
// MI455X (gfx1250) — matrix operand order repaired and checked by reading, not compiled, not run on hardware
//
#include <hip/hip_runtime.h>
#include <hip/hip_bf16.h>

typedef __attribute__((ext_vector_type(16))) _Float16 v16h;
typedef __attribute__((ext_vector_type(8)))  float    v8f;

// ---------------------------------------------------------------------------
// Generic float zero-fill
// ---------------------------------------------------------------------------
__global__ void k_zero(float* __restrict__ p, long n) {
    long t = (long)blockIdx.x * blockDim.x + threadIdx.x;
    if (t < n) p[t] = 0.0f;
}

// deg[dst] += 1 per edge
__global__ void k_degree(const int* __restrict__ ei, float* __restrict__ deg, int E) {
    int e = blockIdx.x * blockDim.x + threadIdx.x;
    if (e < E) atomicAdd(deg + ei[E + e], 1.0f);
}

// cnt[batch[n]] += 1 per node
__global__ void k_counts(const int* __restrict__ batch, float* __restrict__ cnt, int N) {
    int n = blockIdx.x * blockDim.x + threadIdx.x;
    if (n < N) atomicAdd(cnt + batch[n], 1.0f);
}

// dinv = rsqrt(deg + 1) in place
__global__ void k_rsqrt(float* __restrict__ d, int N) {
    int n = blockIdx.x * blockDim.x + threadIdx.x;
    if (n < N) d[n] = rsqrtf(d[n] + 1.0f);
}

// ---------------------------------------------------------------------------
// WMMA GEMM: out[N x 128] = in[N x 128] @ W[128 x 128]   (f16 mul, f32 acc)
// Block = 128 threads (4 waves). Each wave computes a 16x128 output strip.
// W is staged once per block into LDS, pre-swizzled into the documented
// 16-bit B-fragment layout so each lane reads its v16h with one aligned load.
// ---------------------------------------------------------------------------
__global__ __launch_bounds__(128) void k_gemm128(
    const float* __restrict__ in, const float* __restrict__ W,
    float* __restrict__ out, int nrows)
{
    // B fragment store: element index e = ((kstep*8 + nt)*32 + lane)*16 + i
    // value = (f16) W[kstep*32 + (lane>>4)*8 + (i&7) + (i>>3)*16][nt*16 + (lane&15)]
    __shared__ __attribute__((aligned(32))) _Float16 wlds[128 * 128];
    for (int e = threadIdx.x; e < 128 * 128; e += 128) {
        int frag   = e >> 9;          // 0..31
        int within = e & 511;
        int lw     = within >> 4;     // lane 0..31
        int i      = within & 15;
        int kstep  = frag >> 3;
        int nt     = frag & 7;
        int k      = kstep * 32 + (lw >> 4) * 8 + (i & 7) + (i >> 3) * 16;
        int n      = nt * 16 + (lw & 15);
        wlds[e] = (_Float16)W[k * 128 + n];
    }
    __syncthreads();

    const int wave = threadIdx.x >> 5;
    const int lane = threadIdx.x & 31;
    const int rowBase = (blockIdx.x * 4 + wave) * 16;
    if (rowBase >= nrows) return;           // wave-uniform: EXEC stays all-ones

    const int half = lane >> 4;
    const int l15  = lane & 15;
    int m = rowBase + l15;
    if (m >= nrows) m = nrows - 1;          // tail safety (N%16==0 in practice)
    const float* __restrict__ row = in + (size_t)m * 128;

    v8f acc[8] = {};

#pragma unroll
    for (int kstep = 0; kstep < 4; ++kstep) {
        const int k0 = kstep * 32;
        // A fragment, 16-bit 16x32 layout:
        // lanes 0-15 : K = k0+{0..7, 16..23}; lanes 16-31: K = k0+{8..15, 24..31}
        const float4 c0 = *(const float4*)(row + k0 + half * 8);
        const float4 c1 = *(const float4*)(row + k0 + half * 8 + 4);
        const float4 c2 = *(const float4*)(row + k0 + 16 + half * 8);
        const float4 c3 = *(const float4*)(row + k0 + 16 + half * 8 + 4);
        v16h a;
        a[0]  = (_Float16)c0.x; a[1]  = (_Float16)c0.y;
        a[2]  = (_Float16)c0.z; a[3]  = (_Float16)c0.w;
        a[4]  = (_Float16)c1.x; a[5]  = (_Float16)c1.y;
        a[6]  = (_Float16)c1.z; a[7]  = (_Float16)c1.w;
        a[8]  = (_Float16)c2.x; a[9]  = (_Float16)c2.y;
        a[10] = (_Float16)c2.z; a[11] = (_Float16)c2.w;
        a[12] = (_Float16)c3.x; a[13] = (_Float16)c3.y;
        a[14] = (_Float16)c3.z; a[15] = (_Float16)c3.w;

#pragma unroll
        for (int nt = 0; nt < 8; ++nt) {
            v16h b = *(const v16h*)(&wlds[((kstep * 8 + nt) * 32 + lane) * 16]);
            acc[nt] = __builtin_amdgcn_wmma_f32_16x16x32_f16(
                false, a, false, b, (short)0, acc[nt], false, false);
        }
    }

    // D layout: VGPR r -> row rowBase + half*8 + r, col nt*16 + l15
#pragma unroll
    for (int nt = 0; nt < 8; ++nt) {
        float* __restrict__ op =
            out + (size_t)(rowBase + half * 8) * 128 + nt * 16 + l15;
#pragma unroll
        for (int r = 0; r < 8; ++r) {
            if (rowBase + half * 8 + r < nrows)
                op[(size_t)r * 128] = acc[nt][r];
        }
    }
}

// ---------------------------------------------------------------------------
// Edge aggregation: agg[dst] += dinv[src]*dinv[dst] * hw[src]
// One wave per edge: lane handles a float4 (256B per half-wave, coalesced).
// ---------------------------------------------------------------------------
__global__ void k_agg(const int* __restrict__ ei, const float* __restrict__ dinv,
                      const float* __restrict__ hw, float* __restrict__ agg, int E)
{
    const int lane = threadIdx.x & 31;
    int w  = (int)((blockIdx.x * (long)blockDim.x + threadIdx.x) >> 5);
    int nw = (int)(((long)gridDim.x * blockDim.x) >> 5);
    for (int e = w; e < E; e += nw) {
        const int s = ei[e];
        const int d = ei[E + e];
        const float nrm = dinv[s] * dinv[d];
        const float4 v = *(const float4*)(hw + (size_t)s * 128 + lane * 4);
        float* __restrict__ ap = agg + (size_t)d * 128 + lane * 4;
        atomicAdd(ap + 0, nrm * v.x);
        atomicAdd(ap + 1, nrm * v.y);
        atomicAdd(ap + 2, nrm * v.z);
        atomicAdd(ap + 3, nrm * v.w);
    }
}

// h = (agg + dinv^2 * hw + b), optional ReLU, written in place into agg
__global__ void k_self_bias(const float* __restrict__ hw, float* __restrict__ agg,
                            const float* __restrict__ dinv, const float* __restrict__ b,
                            long total, int relu)
{
    long t = (long)blockIdx.x * blockDim.x + threadIdx.x;
    if (t >= total) return;
    int n = (int)(t >> 7);
    int f = (int)(t & 127);
    float di = dinv[n];
    float v = agg[t] + di * di * hw[t] + b[f];
    if (relu) v = fmaxf(v, 0.0f);
    agg[t] = v;
}

// Final layer epilogue fused with mean-pool accumulation:
// out[batch[n], f] += agg[n,f] + dinv[n]^2*hw[n,f] + b[f]
__global__ void k_final_pool(const float* __restrict__ hw, const float* __restrict__ agg,
                             const float* __restrict__ dinv, const float* __restrict__ b,
                             const int* __restrict__ batch, float* __restrict__ out,
                             long total)
{
    long t = (long)blockIdx.x * blockDim.x + threadIdx.x;
    if (t >= total) return;
    int n = (int)(t >> 7);
    int f = (int)(t & 127);
    float di = dinv[n];
    float v = agg[t] + di * di * hw[t] + b[f];
    atomicAdd(out + (size_t)batch[n] * 128 + f, v);
}

__global__ void k_div(float* __restrict__ out, const float* __restrict__ cnt, long total)
{
    long t = (long)blockIdx.x * blockDim.x + threadIdx.x;
    if (t >= total) return;
    out[t] = out[t] / fmaxf(cnt[t >> 7], 1.0f);
}

// ---------------------------------------------------------------------------
extern "C" void kernel_launch(void* const* d_in, const int* in_sizes, int n_in,
                              void* d_out, int out_size, void* d_ws, size_t ws_size,
                              hipStream_t stream)
{
    const float* x     = (const float*)d_in[0];
    const int*   ei    = (const int*)  d_in[1];   // [2, E] src;dst
    const int*   batch = (const int*)  d_in[2];
    const float* W1    = (const float*)d_in[3];
    const float* b1    = (const float*)d_in[4];
    const float* W2    = (const float*)d_in[5];
    const float* b2    = (const float*)d_in[6];

    const int N = in_sizes[0] / 128;
    const int E = in_sizes[1] / 2;
    const int G = out_size / 128;
    const long NF = (long)N * 128;
    const long GF = (long)G * 128;

    float* ws   = (float*)d_ws;
    float* deg  = ws;                         // [N] -> becomes dinv
    float* cnt  = ws + N;                     // [G]
    size_t oA   = ((size_t)(N + G) + 255) & ~(size_t)255;
    float* bufA = ws + oA;                    // [N*128] hw (both layers)
    float* bufB = bufA + (size_t)N * 128;     // [N*128] agg -> h
    float* out  = (float*)d_out;

    const int T = 256;
    auto blk = [](long n, int t) { return (int)((n + t - 1) / t); };

    // ---- degrees / counts / output init ----
    k_zero<<<blk(N, T), T, 0, stream>>>(deg, N);
    k_zero<<<blk(G, T), T, 0, stream>>>(cnt, G);
    k_zero<<<blk(GF, T), T, 0, stream>>>(out, GF);
    k_degree<<<blk(E, T), T, 0, stream>>>(ei, deg, E);
    k_counts<<<blk(N, T), T, 0, stream>>>(batch, cnt, N);
    k_rsqrt<<<blk(N, T), T, 0, stream>>>(deg, N);        // deg -> dinv

    const int rows16     = (N + 15) / 16;
    const int gemmBlocks = (rows16 + 3) / 4;
    const int aggBlocks  = (int)(((long)E * 32 + T - 1) / T);

    // ---- layer 1 ----
    k_gemm128<<<gemmBlocks, 128, 0, stream>>>(x, W1, bufA, N);
    k_zero<<<blk(NF, T), T, 0, stream>>>(bufB, NF);
    k_agg<<<aggBlocks, T, 0, stream>>>(ei, deg, bufA, bufB, E);
    k_self_bias<<<blk(NF, T), T, 0, stream>>>(bufA, bufB, deg, b1, NF, 1);

    // ---- layer 2 ----
    k_gemm128<<<gemmBlocks, 128, 0, stream>>>(bufB, W2, bufA, N);
    k_zero<<<blk(NF, T), T, 0, stream>>>(bufB, NF);
    k_agg<<<aggBlocks, T, 0, stream>>>(ei, deg, bufA, bufB, E);

    // ---- epilogue + mean pool ----
    k_final_pool<<<blk(NF, T), T, 0, stream>>>(bufA, bufB, deg, b2, batch, out, NF);
    k_div<<<blk(GF, T), T, 0, stream>>>(out, cnt, GF);
}
